// DGCNNPartSeg_63015760167487
// MI455X (gfx1250) — compile-verified, matrix operand order checked by reading, not run on hardware
//
#include <hip/hip_runtime.h>
#include <hip/hip_fp16.h>

typedef __attribute__((ext_vector_type(16))) _Float16 v16h;
typedef __attribute__((ext_vector_type(8)))  _Float16 v8h_t;
typedef __attribute__((ext_vector_type(4)))  _Float16 v4h_t;
typedef __attribute__((ext_vector_type(8)))  float    v8f;
typedef __attribute__((ext_vector_type(4)))  float    f4_t;
typedef int v4i_vs __attribute__((vector_size(4 * sizeof(int))));

#define BB   8
#define NN   2048
#define KNN  40
#define NK   (NN * KNN)   // 81920
#define NCLS 16
#define NSEG 50

#ifndef __has_builtin
#define __has_builtin(x) 0
#endif
#if __has_builtin(__builtin_amdgcn_global_load_async_to_lds_b128)
#define HAVE_ASYNC 1
#else
#define HAVE_ASYNC 0
#endif

#define AS1 __attribute__((address_space(1)))
#define AS3 __attribute__((address_space(3)))

// 16B global->LDS copy: async (ASYNCcnt) when available, b128 via VGPRs otherwise
__device__ __forceinline__ void async_ld_b128(_Float16* lds, const _Float16* g) {
#if HAVE_ASYNC
    v4i_vs* gv = (v4i_vs*)g;      // drop const, retype (generic AS)
    v4i_vs* lv = (v4i_vs*)lds;
    __builtin_amdgcn_global_load_async_to_lds_b128(
        (AS1 v4i_vs*)gv, (AS3 v4i_vs*)lv, 0, 0);
#else
    *reinterpret_cast<v8h_t*>(lds) = *reinterpret_cast<const v8h_t*>(g);
#endif
}
__device__ __forceinline__ void async_wait() {
#if HAVE_ASYNC
#if __has_builtin(__builtin_amdgcn_s_wait_asynccnt)
    __builtin_amdgcn_s_wait_asynccnt(0);
#else
    asm volatile("s_wait_asynccnt 0" ::: "memory");
#endif
#endif
}

// ---------------------------------------------------------------------------
// WMMA fragment build: tiles stored [row][40] halves; per lane two aligned
// b128 LDS loads (K-split {0..7,16..23} / {8..15,24..31} per ISA 7.12.2).
// ---------------------------------------------------------------------------
__device__ __forceinline__ v16h make_frag(const _Float16* tile, int row) {
    const int lane = threadIdx.x & 31;
    const int klo  = (lane & 16) ? 8 : 0;
    const v8h_t a0 = *reinterpret_cast<const v8h_t*>(tile + row * 40 + klo);
    const v8h_t a1 = *reinterpret_cast<const v8h_t*>(tile + row * 40 + klo + 16);
    return __builtin_shufflevector(a0, a1, 0,1,2,3,4,5,6,7,8,9,10,11,12,13,14,15);
}

__device__ __forceinline__ float lrelu(float v) { return v > 0.f ? v : 0.2f * v; }

// f32 -> f16 weight conversion (once per launch; makes weight tiles async-able)
__global__ void wcvt_kernel(const float* __restrict__ src, _Float16* __restrict__ dst, int n) {
    int t = blockIdx.x * blockDim.x + threadIdx.x;
    if (t < n) dst[t] = (_Float16)src[t];
}

// x (B,3,N) -> x_cl (B,N,3) channel-last
__global__ void xcl_kernel(const float* __restrict__ x, float* __restrict__ xcl) {
    int t = blockIdx.x * blockDim.x + threadIdx.x;
    if (t >= BB * 3 * NN) return;
    int n = t % NN, rem = t / NN;
    int c = rem % 3, b = rem / 3;
    xcl[((size_t)b * NN + n) * 3 + c] = x[((size_t)b * 3 + c) * NN + n];
}

// xx[b][n] = sum_c X[b,n,coff+c]^2  (channel-last, row stride CB)
__global__ void xx_kernel(const float* __restrict__ X, float* __restrict__ xx,
                          int CB, int coff, int C) {
    int t = blockIdx.x * blockDim.x + threadIdx.x;
    if (t >= BB * NN) return;
    float s = 0.f;
    const float* row = X + (size_t)t * CB + coff;
    for (int c = 0; c < C; ++c) { float v = row[c]; s += v * v; }
    xx[t] = s;
}

// ---------------------------------------------------------------------------
// Dist[b][n][m] = 2*(x x^T)[n,m] - xx[n] - xx[m]  (X channel-last) via WMMA
// Two-phase staging: batch all global loads into registers, then LDS stores.
// ---------------------------------------------------------------------------
__global__ void gram_dist_kernel(const float* __restrict__ X, int CB, int coff, int C,
                                 const float* __restrict__ xx, float* __restrict__ Dist) {
    __shared__ __align__(16) _Float16 At[64 * 40];
    __shared__ __align__(16) _Float16 Bt[128 * 40];
    const int b = blockIdx.z;
    const int nbase = blockIdx.y * 64;
    const int mbase = blockIdx.x * 128;
    const int tid = threadIdx.x, lane = tid & 31, wid = tid >> 5;
    const int ow = wid & 3, pw = wid >> 2;
    v8f acc[4] = {};

    for (int cb = 0; cb < C; cb += 32) {
        if ((CB & 3) == 0) {            // 16B-aligned rows: vector path
            f4_t va[2], vb[4];
#pragma unroll
            for (int r = 0; r < 2; ++r) {                    // A loads (batched)
                int q = tid + 256 * r, row = q >> 3, g = q & 7;
                va[r] = *reinterpret_cast<const f4_t*>(
                    X + ((size_t)b * NN + nbase + row) * CB + coff + cb + g * 4);
            }
#pragma unroll
            for (int r = 0; r < 4; ++r) {                    // B loads (batched)
                int q = tid + 256 * r, col = q >> 3, g = q & 7;
                vb[r] = *reinterpret_cast<const f4_t*>(
                    X + ((size_t)b * NN + mbase + col) * CB + coff + cb + g * 4);
            }
#pragma unroll
            for (int r = 0; r < 2; ++r) {                    // A stores
                int q = tid + 256 * r, row = q >> 3, g = q & 7;
                v4h_t pk;
#pragma unroll
                for (int e = 0; e < 4; ++e)
                    pk[e] = (_Float16)((cb + g * 4 + e < C) ? va[r][e] : 0.f);
                *reinterpret_cast<v4h_t*>(&At[row * 40 + g * 4]) = pk;
            }
#pragma unroll
            for (int r = 0; r < 4; ++r) {                    // B stores
                int q = tid + 256 * r, col = q >> 3, g = q & 7;
                v4h_t pk;
#pragma unroll
                for (int e = 0; e < 4; ++e)
                    pk[e] = (_Float16)((cb + g * 4 + e < C) ? vb[r][e] : 0.f);
                *reinterpret_cast<v4h_t*>(&Bt[col * 40 + g * 4]) = pk;
            }
        } else {                        // C=3 points: batched scalar path
            float va[8], vb[16];
#pragma unroll
            for (int r = 0; r < 8; ++r) {
                int q = tid + 256 * r, row = q >> 5, c = q & 31, cc = cb + c;
                int ccl = (cc < C) ? cc : 0;
                va[r] = X[((size_t)b * NN + nbase + row) * CB + coff + ccl];
            }
#pragma unroll
            for (int r = 0; r < 16; ++r) {
                int q = tid + 256 * r, col = q & 127, c = q >> 7, cc = cb + c;
                int ccl = (cc < C) ? cc : 0;
                vb[r] = X[((size_t)b * NN + mbase + col) * CB + coff + ccl];
            }
#pragma unroll
            for (int r = 0; r < 8; ++r) {
                int q = tid + 256 * r, row = q >> 5, c = q & 31, cc = cb + c;
                At[row * 40 + c] = (_Float16)((cc < C) ? va[r] : 0.f);
            }
#pragma unroll
            for (int r = 0; r < 16; ++r) {
                int q = tid + 256 * r, col = q & 127, c = q >> 7, cc = cb + c;
                Bt[col * 40 + c] = (_Float16)((cc < C) ? vb[r] : 0.f);
            }
        }
        __syncthreads();
        v16h a = make_frag(At, ow * 16 + (lane & 15));
#pragma unroll
        for (int j = 0; j < 4; ++j) {
            v16h bf = make_frag(Bt, pw * 64 + j * 16 + (lane & 15));
            acc[j] = __builtin_amdgcn_wmma_f32_16x16x32_f16(
                false, a, false, bf, (short)0, acc[j], false, false);
        }
        __syncthreads();
    }
#pragma unroll
    for (int j = 0; j < 4; ++j) {
        int m = mbase + pw * 64 + j * 16 + (lane & 15);
        float xxm = xx[b * NN + m];
#pragma unroll
        for (int e = 0; e < 8; ++e) {
            int n = nbase + ow * 16 + ((lane & 16) ? 8 : 0) + e;
            Dist[((size_t)b * NN + n) * NN + m] = 2.f * acc[j][e] - xx[b * NN + n] - xxm;
        }
    }
}

// ---------------------------------------------------------------------------
// top-40: one wave32 per point, 64 dists/lane in regs, shfl_xor reductions
// ---------------------------------------------------------------------------
__global__ void topk_kernel(const float* __restrict__ Dist, int* __restrict__ idxOut) {
    const int lane = threadIdx.x & 31, wid = threadIdx.x >> 5;
    const int pn = blockIdx.x * 8 + wid;
    const int b = pn >> 11, n = pn & (NN - 1);
    const float* row = Dist + ((size_t)b * NN + n) * NN;
    float d[64];
#pragma unroll
    for (int j = 0; j < 64; ++j) d[j] = row[lane + (j << 5)];
    for (int kk = 0; kk < KNN; ++kk) {
        float bv = -1e30f; int bi = 0;
#pragma unroll
        for (int j = 0; j < 64; ++j) {
            int m = lane + (j << 5);
            if (d[j] > bv) { bv = d[j]; bi = m; }
        }
#pragma unroll
        for (int s = 16; s > 0; s >>= 1) {
            float ov = __shfl_xor(bv, s, 32);
            int   oi = __shfl_xor(bi, s, 32);
            if (ov > bv || (ov == bv && oi < bi)) { bv = ov; bi = oi; }
        }
        if (lane == 0) idxOut[((size_t)b * NN + n) * KNN + kk] = bi;
#pragma unroll
        for (int j = 0; j < 64; ++j)
            if ((lane + (j << 5)) == bi) d[j] = -1e30f;
    }
}

// ---------------------------------------------------------------------------
// Fused gather + edge-conv layer 1 (O=64), channel-last f16 output (B,NK,64)
// Two-phase staging: 32 batched gather loads per thread, then LDS stores.
// ---------------------------------------------------------------------------
__global__ void edge_gather_gemm_kernel(const float* __restrict__ X, int CB, int coff, int C,
                                        const int* __restrict__ idx,
                                        const float* __restrict__ W,
                                        _Float16* __restrict__ OUT) {
    __shared__ __align__(16) _Float16 Wt[64 * 40];
    __shared__ __align__(16) _Float16 Xt[128 * 40];
    __shared__ int sN[128], sNb[128];
    const int b = blockIdx.z;
    const int pbase = blockIdx.x * 128;
    const int tid = threadIdx.x, lane = tid & 31, wid = tid >> 5;
    const int ow = wid & 3, pw = wid >> 2;
    const int C2 = 2 * C;
    v8f acc[4] = {};

    if (tid < 128) {
        int p = pbase + tid;
        int n = p / KNN, k = p - n * KNN;
        sN[tid]  = n;
        sNb[tid] = idx[((size_t)b * NN + n) * KNN + k];
    }
    for (int cb = 0; cb < C2; cb += 32) {
        __syncthreads();
        float wv[8], vc[16], vn[16];
#pragma unroll
        for (int r = 0; r < 8; ++r) {                        // weight loads (batched)
            int q = tid + 256 * r, o = q >> 5, c = q & 31, cc = cb + c;
            int ccl = (cc < C2) ? cc : (C2 - 1);
            wv[r] = W[o * C2 + ccl];
        }
#pragma unroll
        for (int r = 0; r < 16; ++r) {                       // gather loads (batched)
            int q = tid + 256 * r, col = q & 127, c = q >> 7, cc = cb + c;
            int n = sN[col], nb = sNb[col];
            int ccl = (cc < C) ? cc : (cc - C);
            ccl = (cc < C2) ? ccl : 0;
            vc[r] = X[((size_t)b * NN + n)  * CB + coff + ccl];
            vn[r] = X[((size_t)b * NN + nb) * CB + coff + ccl];
        }
#pragma unroll
        for (int r = 0; r < 8; ++r) {                        // weight stores
            int q = tid + 256 * r, o = q >> 5, c = q & 31, cc = cb + c;
            Wt[o * 40 + c] = (_Float16)((cc < C2) ? wv[r] : 0.f);
        }
#pragma unroll
        for (int r = 0; r < 16; ++r) {                       // feature stores
            int q = tid + 256 * r, col = q & 127, c = q >> 7, cc = cb + c;
            float v = (cc < C) ? (vn[r] - vc[r]) : vc[r];
            Xt[col * 40 + c] = (_Float16)((cc < C2) ? v : 0.f);
        }
        __syncthreads();
        v16h a = make_frag(Wt, ow * 16 + (lane & 15));
#pragma unroll
        for (int j = 0; j < 4; ++j) {
            v16h bf = make_frag(Xt, pw * 64 + j * 16 + (lane & 15));
            acc[j] = __builtin_amdgcn_wmma_f32_16x16x32_f16(
                false, a, false, bf, (short)0, acc[j], false, false);
        }
    }
#pragma unroll
    for (int j = 0; j < 4; ++j) {                            // packed b128 stores
        int p = pbase + pw * 64 + j * 16 + (lane & 15);
        int cbase = ow * 16 + ((lane & 16) ? 8 : 0);
        v8h_t pk;
#pragma unroll
        for (int e = 0; e < 8; ++e) pk[e] = (_Float16)lrelu(acc[j][e]);
        *reinterpret_cast<v8h_t*>(&OUT[((size_t)b * NK + p) * 64 + cbase]) = pk;
    }
}

// ---------------------------------------------------------------------------
// Generic 1x1 conv GEMM, channel-last IO. IN f16 path: async b128 -> LDS.
// OUT f16: packed channel-last; OUT float: channel-major (final layer only).
// ---------------------------------------------------------------------------
template <typename TIn, typename TOut, bool LRELU>
__global__ void gemm_kernel(const TIn* __restrict__ IN, const _Float16* __restrict__ W16,
                            TOut* __restrict__ OUT, int O, int Cin, int P) {
    __shared__ __align__(16) _Float16 Wt[64 * 40];
    __shared__ __align__(16) _Float16 Xt[128 * 40];
    const int b = blockIdx.z;
    const int obase = blockIdx.y * 64;
    const int pbase = blockIdx.x * 128;
    const int tid = threadIdx.x, lane = tid & 31, wid = tid >> 5;
    const int ow = wid & 3, pw = wid >> 2;
    v8f acc[4] = {};

    for (int cb = 0; cb < Cin; cb += 32) {
        {   // weight tile: 64 rows x 32ch = 256 x 16B async
            int r = tid >> 2, sub = tid & 3;
            async_ld_b128(&Wt[r * 40 + sub * 8],
                          W16 + ((size_t)(obase + r)) * Cin + cb + sub * 8);
        }
        if constexpr (sizeof(TIn) == 2) {   // f16 input: 512 x 16B async
#pragma unroll
            for (int r = 0; r < 2; ++r) {
                int q = tid + 256 * r, col = q >> 2, sub = q & 3;
                async_ld_b128(&Xt[col * 40 + sub * 8],
                              (const _Float16*)IN + ((size_t)b * P + pbase + col) * Cin + cb + sub * 8);
            }
        } else {                            // f32 input: batched vector cvt path
            f4_t vv[4];
#pragma unroll
            for (int r = 0; r < 4; ++r) {
                int q = tid + 256 * r, col = q >> 3, g = q & 7;
                vv[r] = *reinterpret_cast<const f4_t*>(
                    (const float*)IN + ((size_t)b * P + pbase + col) * Cin + cb + g * 4);
            }
#pragma unroll
            for (int r = 0; r < 4; ++r) {
                int q = tid + 256 * r, col = q >> 3, g = q & 7;
                v4h_t pk;
#pragma unroll
                for (int e = 0; e < 4; ++e) pk[e] = (_Float16)vv[r][e];
                *reinterpret_cast<v4h_t*>(&Xt[col * 40 + g * 4]) = pk;
            }
        }
        async_wait();
        __syncthreads();
        v16h a = make_frag(Wt, ow * 16 + (lane & 15));
#pragma unroll
        for (int j = 0; j < 4; ++j) {
            v16h bf = make_frag(Xt, pw * 64 + j * 16 + (lane & 15));
            acc[j] = __builtin_amdgcn_wmma_f32_16x16x32_f16(
                false, a, false, bf, (short)0, acc[j], false, false);
        }
        __syncthreads();
    }
#pragma unroll
    for (int j = 0; j < 4; ++j) {
        int p = pbase + pw * 64 + j * 16 + (lane & 15);
        int cbase = obase + ow * 16 + ((lane & 16) ? 8 : 0);
        if constexpr (sizeof(TOut) == 2) {  // channel-last packed b128
            v8h_t pk;
#pragma unroll
            for (int e = 0; e < 8; ++e) {
                float v = acc[j][e];
                if (LRELU) v = lrelu(v);
                pk[e] = (_Float16)v;
            }
            *reinterpret_cast<v8h_t*>(&((_Float16*)OUT)[((size_t)b * P + p) * O + cbase]) = pk;
        } else {                            // channel-major f32 (final layer)
#pragma unroll
            for (int e = 0; e < 8; ++e) {
                int o = cbase + e;
                if (o < O) {
                    float v = acc[j][e];
                    if (LRELU) v = lrelu(v);
                    ((float*)OUT)[((size_t)b * O + o) * P + p] = v;
                }
            }
        }
    }
}

// ---------------------------------------------------------------------------
// max over K: H (B,NK,O) f16 -> OUT[(b*N+n)*OST + coff + o], 8-ch vector
// ---------------------------------------------------------------------------
template <typename TOut>
__global__ void maxk_kernel(const _Float16* __restrict__ H, TOut* __restrict__ OUT,
                            int O, int OST, int coff) {
    int t = blockIdx.x * blockDim.x + threadIdx.x;
    int og8 = O >> 3;
    if (t >= BB * NN * og8) return;
    int og = t % og8, rem = t / og8;
    int n = rem % NN, b = rem / NN;
    const _Float16* src = H + ((size_t)b * NK + (size_t)n * KNN) * O + og * 8;
    float m[8];
#pragma unroll
    for (int e = 0; e < 8; ++e) m[e] = -1e30f;
    for (int k = 0; k < KNN; ++k) {
        v8h_t v = *reinterpret_cast<const v8h_t*>(src + (size_t)k * O);
#pragma unroll
        for (int e = 0; e < 8; ++e) m[e] = fmaxf(m[e], (float)v[e]);
    }
    TOut* dst = OUT + ((size_t)b * NN + n) * OST + coff + og * 8;
#pragma unroll
    for (int e = 0; e < 8; ++e) dst[e] = (TOut)m[e];
}

// max over N: in (B,N,O) f16 -> out (B,O) f32, 8-ch vector per thread
__global__ void maxn_kernel(const _Float16* __restrict__ in, float* __restrict__ out, int O) {
    int t = blockIdx.x * blockDim.x + threadIdx.x;
    int og8 = O >> 3;
    if (t >= BB * og8) return;
    int og = t % og8, b = t / og8;
    float m[8];
#pragma unroll
    for (int e = 0; e < 8; ++e) m[e] = -1e30f;
    for (int n = 0; n < NN; ++n) {
        v8h_t v = *reinterpret_cast<const v8h_t*>(in + ((size_t)b * NN + n) * O + og * 8);
#pragma unroll
        for (int e = 0; e < 8; ++e) m[e] = fmaxf(m[e], (float)v[e]);
    }
#pragma unroll
    for (int e = 0; e < 8; ++e) out[b * O + og * 8 + e] = m[e];
}

// small FC: out[r][o] = act(sum_c in[r][c]*W[o][c] + bias[o])
__global__ void fc_kernel(const float* __restrict__ in, const float* __restrict__ W,
                          const float* __restrict__ bias, float* __restrict__ out,
                          int O, int C, int rows, int doLrelu) {
    int t = blockIdx.x * blockDim.x + threadIdx.x;
    if (t >= rows * O) return;
    int r = t / O, o = t % O;
    float s = bias ? bias[o] : 0.f;
    for (int c = 0; c < C; ++c) s += in[r * C + c] * W[o * C + c];
    out[t] = doLrelu ? lrelu(s) : s;
}

// xT[(b*N+n)*3+i] = sum_j T9[b][j*3+i] * x[b][j][n]   (x channel-major input)
__global__ void transform_kernel(const float* __restrict__ T9, const float* __restrict__ x,
                                 float* __restrict__ xT) {
    int t = blockIdx.x * blockDim.x + threadIdx.x;
    if (t >= BB * 3 * NN) return;
    int i = t % 3, rem = t / 3;
    int n = rem % NN, b = rem / NN;
    float s = 0.f;
#pragma unroll
    for (int j = 0; j < 3; ++j)
        s += T9[b * 9 + j * 3 + i] * x[((size_t)b * 3 + j) * NN + n];
    xT[t] = s;
}

// hcat (B,N,1280) f16 = [g(1024) bcast ; lv(64) bcast ; x1,x2,x3 (192)]
__global__ void hcat_kernel(const float* __restrict__ g, const float* __restrict__ lv,
                            const float* __restrict__ X123, _Float16* __restrict__ hcat) {
    size_t t = (size_t)blockIdx.x * blockDim.x + threadIdx.x;
    if (t >= (size_t)BB * NN * 1280) return;
    int c = t % 1280; size_t rem = t / 1280;
    int n = rem % NN, b = rem / NN;
    float v;
    if (c < 1024)      v = g[b * 1024 + c];
    else if (c < 1088) v = lv[b * 64 + (c - 1024)];
    else               v = X123[((size_t)b * NN + n) * 192 + (c - 1088)];
    hcat[t] = (_Float16)v;
}

// ---------------------------------------------------------------------------
extern "C" void kernel_launch(void* const* d_in, const int* in_sizes, int n_in,
                              void* d_out, int out_size, void* d_ws, size_t ws_size,
                              hipStream_t stream) {
    (void)in_sizes; (void)n_in; (void)out_size; (void)ws_size;
    const float* x     = (const float*)d_in[0];
    const float* l     = (const float*)d_in[1];
    const float* t_c1  = (const float*)d_in[2];
    const float* t_c2  = (const float*)d_in[3];
    const float* t_c3  = (const float*)d_in[4];
    const float* t_f1  = (const float*)d_in[5];
    const float* t_f2  = (const float*)d_in[6];
    const float* t_f3w = (const float*)d_in[7];
    const float* t_f3b = (const float*)d_in[8];
    const float* b1a   = (const float*)d_in[9];
    const float* b1b   = (const float*)d_in[10];
    const float* b2a   = (const float*)d_in[11];
    const float* b2b   = (const float*)d_in[12];
    const float* b3a   = (const float*)d_in[13];
    const float* m1    = (const float*)d_in[14];
    const float* m2    = (const float*)d_in[15];
    const float* h1    = (const float*)d_in[16];
    const float* h2    = (const float*)d_in[17];
    const float* h3    = (const float*)d_in[18];
    const float* h4    = (const float*)d_in[19];
    float* out = (float*)d_out;

    char* cur = (char*)d_ws;
    auto take = [&](size_t bytes) { char* p = cur; cur += (bytes + 255) & ~(size_t)255; return p; };
    char* R1 = take(176160768ull);                  // max(dist 134MB, H2 168MB)
    char* R2 = take(88080384ull);                   // H1 84MB / hcat + head bufs
    float*    dist  = (float*)R1;
    _Float16* H2b   = (_Float16*)R1;
    _Float16* H1b   = (_Float16*)R2;
    _Float16* hcat  = (_Float16*)R2;
    _Float16* Hh1   = (_Float16*)(R2 + 44040192ull);
    _Float16* Hh2   = (_Float16*)(R2 + 53477376ull);
    _Float16* Hh3   = (_Float16*)(R2 + 62914560ull);
    int*      idxb  = (int*)take((size_t)BB * NN * KNN * 4);
    float*    xxb   = (float*)take((size_t)BB * NN * 4);
    _Float16* Tfeat = (_Float16*)take((size_t)BB * NN * 128 * 2);  // (B,N,128) f16
    _Float16* TgG   = (_Float16*)take((size_t)BB * NN * 1024 * 2); // (B,N,1024) f16
    float*    tvec  = (float*)take(BB * 1024 * 4);
    float*    tf1o  = (float*)take(BB * 512 * 4);
    float*    tf2o  = (float*)take(BB * 256 * 4);
    float*    T9    = (float*)take(BB * 9 * 4);
    float*    xcl   = (float*)take((size_t)BB * NN * 3 * 4);
    float*    xT    = (float*)take((size_t)BB * NN * 3 * 4);
    float*    X123  = (float*)take((size_t)BB * NN * 192 * 4);
    float*    gvec  = (float*)take(BB * 1024 * 4);
    float*    lvb   = (float*)take(BB * 64 * 4);
    // f16 weights (rows padded to 64 for unguarded async staging)
    _Float16* w_tc2 = (_Float16*)take(128 * 64 * 2);
    _Float16* w_tc3 = (_Float16*)take(1024 * 128 * 2);
    _Float16* w_b1b = (_Float16*)take(64 * 64 * 2);
    _Float16* w_b2b = (_Float16*)take(64 * 64 * 2);
    _Float16* w_m1  = (_Float16*)take(1024 * 192 * 2);
    _Float16* w_h1  = (_Float16*)take(256 * 1280 * 2);
    _Float16* w_h2  = (_Float16*)take(256 * 256 * 2);
    _Float16* w_h3  = (_Float16*)take(128 * 256 * 2);
    _Float16* w_h4  = (_Float16*)take(64 * 128 * 2);  // 50 rows used
    take(4096);                                        // trailing slack

    const dim3 blk(256);
    const dim3 gGram(NN / 128, NN / 64, BB);
    const dim3 gNK(NK / 128, 1, BB);

    auto cvt = [&](const float* s, _Float16* d, int n) {
        wcvt_kernel<<<(n + 255) / 256, blk, 0, stream>>>(s, d, n);
    };
    cvt(t_c2, w_tc2, 128 * 64);   cvt(t_c3, w_tc3, 1024 * 128);
    cvt(b1b, w_b1b, 64 * 64);     cvt(b2b, w_b2b, 64 * 64);
    cvt(m1, w_m1, 1024 * 192);    cvt(h1, w_h1, 256 * 1280);
    cvt(h2, w_h2, 256 * 256);     cvt(h3, w_h3, 128 * 256);
    cvt(h4, w_h4, 50 * 128);
    xcl_kernel<<<(BB * 3 * NN) / 256, blk, 0, stream>>>(x, xcl);

    // ================= transform net path (original x) ======================
    xx_kernel<<<64, blk, 0, stream>>>(xcl, xxb, 3, 0, 3);
    gram_dist_kernel<<<gGram, blk, 0, stream>>>(xcl, 3, 0, 3, xxb, dist);
    topk_kernel<<<BB * NN / 8, blk, 0, stream>>>(dist, idxb);
    edge_gather_gemm_kernel<<<gNK, blk, 0, stream>>>(xcl, 3, 0, 3, idxb, t_c1, H1b);
    gemm_kernel<_Float16, _Float16, true><<<dim3(NK / 128, 2, BB), blk, 0, stream>>>(
        H1b, w_tc2, H2b, 128, 64, NK);
    maxk_kernel<_Float16><<<BB * NN * 16 / 256, blk, 0, stream>>>(H2b, Tfeat, 128, 128, 0);
    gemm_kernel<_Float16, _Float16, true><<<dim3(NN / 128, 16, BB), blk, 0, stream>>>(
        Tfeat, w_tc3, TgG, 1024, 128, NN);
    maxn_kernel<<<(BB * 128 + 255) / 256, blk, 0, stream>>>(TgG, tvec, 1024);
    fc_kernel<<<(BB * 512 + 255) / 256, blk, 0, stream>>>(tvec, t_f1, nullptr, tf1o, 512, 1024, BB, 1);
    fc_kernel<<<(BB * 256 + 255) / 256, blk, 0, stream>>>(tf1o, t_f2, nullptr, tf2o, 256, 512, BB, 1);
    fc_kernel<<<1, blk, 0, stream>>>(tf2o, t_f3w, t_f3b, T9, 9, 256, BB, 0);
    transform_kernel<<<(BB * 3 * NN) / 256, blk, 0, stream>>>(T9, x, xT);

    // ============================ edge block 1 ==============================
    xx_kernel<<<64, blk, 0, stream>>>(xT, xxb, 3, 0, 3);
    gram_dist_kernel<<<gGram, blk, 0, stream>>>(xT, 3, 0, 3, xxb, dist);
    topk_kernel<<<BB * NN / 8, blk, 0, stream>>>(dist, idxb);
    edge_gather_gemm_kernel<<<gNK, blk, 0, stream>>>(xT, 3, 0, 3, idxb, b1a, H1b);
    gemm_kernel<_Float16, _Float16, true><<<gNK, blk, 0, stream>>>(H1b, w_b1b, H2b, 64, 64, NK);
    maxk_kernel<float><<<BB * NN * 8 / 256, blk, 0, stream>>>(H2b, X123, 64, 192, 0);

    // ============================ edge block 2 ==============================
    xx_kernel<<<64, blk, 0, stream>>>(X123, xxb, 192, 0, 64);
    gram_dist_kernel<<<gGram, blk, 0, stream>>>(X123, 192, 0, 64, xxb, dist);
    topk_kernel<<<BB * NN / 8, blk, 0, stream>>>(dist, idxb);
    edge_gather_gemm_kernel<<<gNK, blk, 0, stream>>>(X123, 192, 0, 64, idxb, b2a, H1b);
    gemm_kernel<_Float16, _Float16, true><<<gNK, blk, 0, stream>>>(H1b, w_b2b, H2b, 64, 64, NK);
    maxk_kernel<float><<<BB * NN * 8 / 256, blk, 0, stream>>>(H2b, X123, 64, 192, 64);

    // ============================ edge block 3 ==============================
    xx_kernel<<<64, blk, 0, stream>>>(X123, xxb, 192, 64, 64);
    gram_dist_kernel<<<gGram, blk, 0, stream>>>(X123, 192, 64, 64, xxb, dist);
    topk_kernel<<<BB * NN / 8, blk, 0, stream>>>(dist, idxb);
    edge_gather_gemm_kernel<<<gNK, blk, 0, stream>>>(X123, 192, 64, 64, idxb, b3a, H1b);
    maxk_kernel<float><<<BB * NN * 8 / 256, blk, 0, stream>>>(H1b, X123, 64, 192, 128);

    // =============================== head ===================================
    gemm_kernel<float, _Float16, true><<<dim3(NN / 128, 16, BB), blk, 0, stream>>>(
        X123, w_m1, TgG, 1024, 192, NN);
    maxn_kernel<<<(BB * 128 + 255) / 256, blk, 0, stream>>>(TgG, gvec, 1024);
    fc_kernel<<<(BB * 64 + 255) / 256, blk, 0, stream>>>(l, m2, nullptr, lvb, 64, NCLS, BB, 1);
    hcat_kernel<<<(unsigned)(((size_t)BB * NN * 1280 + 255) / 256), blk, 0, stream>>>(
        gvec, lvb, X123, hcat);
    gemm_kernel<_Float16, _Float16, true><<<dim3(NN / 128, 4, BB), blk, 0, stream>>>(
        hcat, w_h1, Hh1, 256, 1280, NN);
    gemm_kernel<_Float16, _Float16, true><<<dim3(NN / 128, 4, BB), blk, 0, stream>>>(
        Hh1, w_h2, Hh2, 256, 256, NN);
    gemm_kernel<_Float16, _Float16, true><<<dim3(NN / 128, 2, BB), blk, 0, stream>>>(
        Hh2, w_h3, Hh3, 128, 256, NN);
    gemm_kernel<_Float16, float, false><<<dim3(NN / 128, 1, BB), blk, 0, stream>>>(
        Hh3, w_h4, out, NSEG, 128, NN);
}
